// ArcticMoE_75780402970675
// MI455X (gfx1250) — matrix operand order repaired and checked by reading, not compiled, not run on hardware
//
#include <hip/hip_runtime.h>
#include <hip/hip_bf16.h>

typedef __bf16 bf16_t;
typedef __bf16 v16bf __attribute__((ext_vector_type(16)));
typedef __bf16 v8bf  __attribute__((ext_vector_type(8)));
typedef float  v8f   __attribute__((ext_vector_type(8)));

#define MODEL_DIM 1024
#define INTER_DIM 4096
#define NUM_EXPERTS 8

#define BM 128
#define BN 128
#define BK 32
#define LDT 40   // padded LDS row stride in bf16 elems (80B: 16B-aligned, conflict-spreading)

// ------------------------------------------------- CDNA5 async helpers ------
// ASYNCcnt-tracked direct global->LDS copy (no VGPR staging). INST_OFFSET is
// added to both the LDS and global addresses (ISA §10 async pseudocode), so a
// single base covers both 16B chunks of a 32B row segment.
__device__ __forceinline__ unsigned lds_off(const void* p) {
    // generic pointer to LDS: low 32 bits are the LDS offset
    return (unsigned)(unsigned long long)p;
}
__device__ __forceinline__ void async_copy32(unsigned ldsOff, unsigned long long gaddr) {
    asm volatile(
        "global_load_async_to_lds_b128 %0, %1, off\n\t"
        "global_load_async_to_lds_b128 %0, %1, off offset:16"
        :: "v"(ldsOff), "v"(gaddr) : "memory");
}
__device__ __forceinline__ void wait_asynccnt0() {
    asm volatile("s_wait_asynccnt 0x0" ::: "memory");
}

// ---------------------------------------------------------------- router ----
__global__ void moe_router(const float* __restrict__ x, const float* __restrict__ gate_w,
                           int* __restrict__ topk, int* __restrict__ counts, int T) {
    const int wid  = threadIdx.x >> 5;
    const int lane = threadIdx.x & 31;
    const int t = blockIdx.x * 8 + wid;
    if (t >= T) return;

    float acc[NUM_EXPERTS];
#pragma unroll
    for (int e = 0; e < NUM_EXPERTS; ++e) acc[e] = 0.f;
    const float* xr = x + (long)t * MODEL_DIM;
    for (int j = 0; j < MODEL_DIM / 32; ++j) {
        float xv = xr[j * 32 + lane];
#pragma unroll
        for (int e = 0; e < NUM_EXPERTS; ++e)
            acc[e] += xv * gate_w[e * MODEL_DIM + j * 32 + lane];
    }
#pragma unroll
    for (int off = 16; off > 0; off >>= 1)
#pragma unroll
        for (int e = 0; e < NUM_EXPERTS; ++e)
            acc[e] += __shfl_xor(acc[e], off, 32);

    if (lane == 0) {
        // softmax is monotone -> top-2 of logits; ties keep lowest index (matches lax.top_k)
        int b0 = 0; float v0 = acc[0];
#pragma unroll
        for (int e = 1; e < NUM_EXPERTS; ++e) if (acc[e] > v0) { v0 = acc[e]; b0 = e; }
        int b1 = 0; float v1 = -3.4e38f;
#pragma unroll
        for (int e = 0; e < NUM_EXPERTS; ++e) if (e != b0 && acc[e] > v1) { v1 = acc[e]; b1 = e; }
        topk[t * 2 + 0] = b0;
        topk[t * 2 + 1] = b1;
        atomicAdd(&counts[b0], 1);
        atomicAdd(&counts[b1], 1);
    }
}

__global__ void moe_scan(const int* __restrict__ counts, int* __restrict__ offsets,
                         int* __restrict__ cursor) {
    if (threadIdx.x == 0) {
        int off = 0;
        for (int e = 0; e < NUM_EXPERTS; ++e) { offsets[e] = off; cursor[e] = off; off += counts[e]; }
    }
}

__global__ void moe_scatter(const int* __restrict__ topk, int* __restrict__ cursor,
                            int* __restrict__ row_token, int nrows) {
    int id = blockIdx.x * blockDim.x + threadIdx.x;
    if (id >= nrows) return;
    int e = topk[id];
    int pos = atomicAdd(&cursor[e], 1);
    row_token[pos] = id >> 1;   // token id
}

// ------------------------------------------------------- fragment helpers ---
// 16-bit A 16x32 frag (ISA 7.12.2): lane l -> row (l&15); kbase=(l>=16)?8:0;
// elems 0..7 = K[kbase..kbase+7], elems 8..15 = K[kbase+16..kbase+23].
__device__ __forceinline__ v16bf load_a_frag(const bf16_t* As, int rowBase, int lane) {
    const int r  = rowBase + (lane & 15);
    const int kb = (lane >> 4) * 8;
    const bf16_t* p = As + r * LDT + kb;
    v8bf lo = *(const v8bf*)p;
    v8bf hi = *(const v8bf*)(p + 16);
    return __builtin_shufflevector(lo, hi, 0,1,2,3,4,5,6,7,8,9,10,11,12,13,14,15);
}
// 16-bit B 32x16 frag: lane l -> col (l&15); kbase=(l>=16)?8:0; elems 0..7 = K[kbase..kbase+7], elems 8..15 = K[kbase+16..kbase+23].
// Bs stored [N][K] so this is contiguous per lane.
__device__ __forceinline__ v16bf load_b_frag(const bf16_t* Bs, int colBase, int lane) {
    const int c  = colBase + (lane & 15);
    const int kb = (lane >> 4) * 8;
    const bf16_t* p = Bs + c * LDT + kb;
    v8bf lo = *(const v8bf*)p;
    v8bf hi = *(const v8bf*)(p + 16);
    return __builtin_shufflevector(lo, hi, 0,1,2,3,4,5,6,7,8,9,10,11,12,13,14,15);
}

// --------------------------------------------------------------- up GEMM ----
// rows of moe_in (gathered x, fp32 -> bf16) @ w_up[e] (fp32 -> bf16) -> h (bf16)
// Double-buffered LDS: one barrier per K-step; next tile fills overlap WMMA.
__launch_bounds__(256)
__global__ void moe_up(const float* __restrict__ x, const float* __restrict__ w_up,
                       bf16_t* __restrict__ h,
                       const int* __restrict__ counts, const int* __restrict__ offsets,
                       const int* __restrict__ row_token) {
    __shared__ __attribute__((aligned(16))) bf16_t As[2][BM * LDT];
    __shared__ __attribute__((aligned(16))) bf16_t Bs[2][BN * LDT];

    const int e = blockIdx.z;
    const int rows = counts[e];
    const int mtile = blockIdx.y;
    if (mtile * BM >= rows) return;
    const int rowG0 = offsets[e] + mtile * BM;
    const int rowsValid = min(BM, rows - mtile * BM);
    const int n0 = blockIdx.x * BN;
    const float* W = w_up + (long)e * MODEL_DIM * INTER_DIM;

    const int tid = threadIdx.x;
    const int lane = tid & 31, wid = tid >> 5;
    const int wm = wid & 3, wn = wid >> 2;   // 4x2 wave grid -> wave tile 32x64

    v8f acc[2][4];
    const v8f vzero = {0.f,0.f,0.f,0.f,0.f,0.f,0.f,0.f};
#pragma unroll
    for (int i = 0; i < 2; ++i)
#pragma unroll
        for (int j = 0; j < 4; ++j) acc[i][j] = vzero;

    const int ar = tid >> 1, ah = tid & 1;   // A loader: row, k-half (16 elems)
    const bool aValid = (ar < rowsValid);
    const int atok = aValid ? row_token[rowG0 + ar] : 0;
    const int bn = tid & 127, bh = tid >> 7; // B loader: col, k-half (16 elems)

    // invalid A rows: zero once in both buffers, never touched again
    if (!aValid) {
        v8bf z = {};
#pragma unroll
        for (int b = 0; b < 2; ++b) {
            *(v8bf*)&As[b][ar * LDT + ah * 16]     = z;
            *(v8bf*)&As[b][ar * LDT + ah * 16 + 8] = z;
        }
    }

    auto load_tiles = [&](int buf, int k0) {
        if (aValid) {  // A tile: gather token row, fp32 -> bf16
            bf16_t tmp[16];
            const float4* src = (const float4*)(x + (long)atok * MODEL_DIM + k0 + ah * 16);
#pragma unroll
            for (int q = 0; q < 4; ++q) {
                float4 v = src[q];
                tmp[q*4+0] = (bf16_t)v.x; tmp[q*4+1] = (bf16_t)v.y;
                tmp[q*4+2] = (bf16_t)v.z; tmp[q*4+3] = (bf16_t)v.w;
            }
            *(v8bf*)&As[buf][ar * LDT + ah * 16]     = *(v8bf*)&tmp[0];
            *(v8bf*)&As[buf][ar * LDT + ah * 16 + 8] = *(v8bf*)&tmp[8];
        }
        {  // B tile: w_up rows k0..k0+31, cols n0..n0+127 -> Bs[N][K]
            bf16_t tmp[16];
            const float* src = W + (long)(k0 + bh * 16) * INTER_DIM + n0 + bn;
#pragma unroll
            for (int kk = 0; kk < 16; ++kk) tmp[kk] = (bf16_t)src[(long)kk * INTER_DIM];
            *(v8bf*)&Bs[buf][bn * LDT + bh * 16]     = *(v8bf*)&tmp[0];
            *(v8bf*)&Bs[buf][bn * LDT + bh * 16 + 8] = *(v8bf*)&tmp[8];
            if (k0 + BK < MODEL_DIM)  // nudge following tile toward L2 (global_prefetch_b8)
                __builtin_prefetch(src + (long)BK * INTER_DIM, 0, 1);
        }
    };

    load_tiles(0, 0);
    const int KT = MODEL_DIM / BK;
    for (int it = 0; it < KT; ++it) {
        const int buf = it & 1;
        __syncthreads();                     // tile[buf] ready; buf^1 free for refill
        if (it + 1 < KT) load_tiles(buf ^ 1, (it + 1) * BK);

        v16bf a[2], b[4];
#pragma unroll
        for (int i = 0; i < 2; ++i) a[i] = load_a_frag(As[buf], wm * 32 + i * 16, lane);
#pragma unroll
        for (int j = 0; j < 4; ++j) b[j] = load_b_frag(Bs[buf], wn * 64 + j * 16, lane);
#pragma unroll
        for (int i = 0; i < 2; ++i)
#pragma unroll
            for (int j = 0; j < 4; ++j)
                acc[i][j] = __builtin_amdgcn_wmma_f32_16x16x32_bf16(
                    false, a[i], false, b[j], (short)0, acc[i][j], false, false);
    }
    // epilogue: C elem (v, lane) -> row = v + (lane>=16)*8, col = lane&15
    const int crow = (lane >> 4) * 8, ccol = lane & 15;
#pragma unroll
    for (int i = 0; i < 2; ++i)
#pragma unroll
        for (int j = 0; j < 4; ++j)
#pragma unroll
            for (int v = 0; v < 8; ++v) {
                int lr = wm * 32 + i * 16 + crow + v;
                if (lr < rowsValid)
                    h[(long)(rowG0 + lr) * INTER_DIM + n0 + wn * 64 + j * 16 + ccol]
                        = (bf16_t)acc[i][j][v];
            }
}

// ------------------------------------------------------------- down GEMM ----
// h (bf16) @ w_down[e] (fp32 -> bf16), combine into out via fp32 hw atomics.
// A tiles stream via GLOBAL_LOAD_ASYNC_TO_LDS_B128 (ASYNCcnt), bypassing VGPRs.
__launch_bounds__(256)
__global__ void moe_down(const bf16_t* __restrict__ h, const float* __restrict__ w_down,
                         float* __restrict__ out,
                         const int* __restrict__ counts, const int* __restrict__ offsets,
                         const int* __restrict__ row_token) {
    __shared__ __attribute__((aligned(16))) bf16_t As[2][BM * LDT];
    __shared__ __attribute__((aligned(16))) bf16_t Bs[2][BN * LDT];

    const int e = blockIdx.z;
    const int rows = counts[e];
    const int mtile = blockIdx.y;
    if (mtile * BM >= rows) return;
    const int rowG0 = offsets[e] + mtile * BM;
    const int rowsValid = min(BM, rows - mtile * BM);
    const int n0 = blockIdx.x * BN;
    const float* W = w_down + (long)e * INTER_DIM * MODEL_DIM;

    const int tid = threadIdx.x;
    const int lane = tid & 31, wid = tid >> 5;
    const int wm = wid & 3, wn = wid >> 2;

    v8f acc[2][4];
    const v8f vzero = {0.f,0.f,0.f,0.f,0.f,0.f,0.f,0.f};
#pragma unroll
    for (int i = 0; i < 2; ++i)
#pragma unroll
        for (int j = 0; j < 4; ++j) acc[i][j] = vzero;

    const int ar = tid >> 1, ah = tid & 1;
    const bool aValid = (ar < rowsValid);
    const bf16_t* aSrc = h + (long)(rowG0 + ar) * INTER_DIM + ah * 16;
    const int bn = tid & 127, bh = tid >> 7;

    if (!aValid) {   // zero once; async fills never touch these rows
        v8bf z = {};
#pragma unroll
        for (int b = 0; b < 2; ++b) {
            *(v8bf*)&As[b][ar * LDT + ah * 16]     = z;
            *(v8bf*)&As[b][ar * LDT + ah * 16 + 8] = z;
        }
    }

    auto load_a_async = [&](int buf, int k0) {
        if (aValid)
            async_copy32(lds_off(&As[buf][ar * LDT + ah * 16]),
                         (unsigned long long)(aSrc + k0));
    };
    auto load_b = [&](int buf, int k0) {
        bf16_t tmp[16];
        const float* src = W + (long)(k0 + bh * 16) * MODEL_DIM + n0 + bn;
#pragma unroll
        for (int kk = 0; kk < 16; ++kk) tmp[kk] = (bf16_t)src[(long)kk * MODEL_DIM];
        *(v8bf*)&Bs[buf][bn * LDT + bh * 16]     = *(v8bf*)&tmp[0];
        *(v8bf*)&Bs[buf][bn * LDT + bh * 16 + 8] = *(v8bf*)&tmp[8];
        if (k0 + BK < INTER_DIM)
            __builtin_prefetch(src + (long)BK * MODEL_DIM, 0, 1);
    };

    load_a_async(0, 0);
    load_b(0, 0);
    const int KT = INTER_DIM / BK;
    for (int it = 0; it < KT; ++it) {
        const int buf = it & 1;
        wait_asynccnt0();                    // my async fills of tile[buf] done
        __syncthreads();                     // everyone's fills visible; buf^1 free
        if (it + 1 < KT) {
            load_a_async(buf ^ 1, (it + 1) * BK);
            load_b(buf ^ 1, (it + 1) * BK);
        }

        v16bf a[2], b[4];
#pragma unroll
        for (int i = 0; i < 2; ++i) a[i] = load_a_frag(As[buf], wm * 32 + i * 16, lane);
#pragma unroll
        for (int j = 0; j < 4; ++j) b[j] = load_b_frag(Bs[buf], wn * 64 + j * 16, lane);
#pragma unroll
        for (int i = 0; i < 2; ++i)
#pragma unroll
            for (int j = 0; j < 4; ++j)
                acc[i][j] = __builtin_amdgcn_wmma_f32_16x16x32_bf16(
                    false, a[i], false, b[j], (short)0, acc[i][j], false, false);
    }
    // combine: each token gets exactly two commutative f32 adds -> deterministic
    const int crow = (lane >> 4) * 8, ccol = lane & 15;
#pragma unroll
    for (int i = 0; i < 2; ++i)
#pragma unroll
        for (int j = 0; j < 4; ++j)
#pragma unroll
            for (int v = 0; v < 8; ++v) {
                int lr = wm * 32 + i * 16 + crow + v;
                if (lr < rowsValid) {
                    int tok = row_token[rowG0 + lr];
                    unsafeAtomicAdd(&out[(long)tok * MODEL_DIM + n0 + wn * 64 + j * 16 + ccol],
                                    acc[i][j][v]);
                }
            }
}

// ---------------------------------------------------------------- launch ----
extern "C" void kernel_launch(void* const* d_in, const int* in_sizes, int n_in,
                              void* d_out, int out_size, void* d_ws, size_t ws_size,
                              hipStream_t stream) {
    const float* x      = (const float*)d_in[0];
    const float* gate_w = (const float*)d_in[1];
    const float* w_up   = (const float*)d_in[2];
    const float* w_down = (const float*)d_in[3];
    float* out = (float*)d_out;

    const int T = in_sizes[0] / MODEL_DIM;   // 2048 tokens
    const int R = T * 2;                     // routed rows (top-2)

    char* ws = (char*)d_ws;
    int* counts    = (int*)(ws + 0);
    int* offsets   = (int*)(ws + 32);
    int* cursor    = (int*)(ws + 64);
    int* topk      = (int*)(ws + 128);
    int* row_token = (int*)(ws + 128 + sizeof(int) * (size_t)R);
    bf16_t* h      = (bf16_t*)(ws + 65536);  // R x INTER_DIM bf16 (32 MB)

    hipMemsetAsync(d_out, 0, (size_t)out_size * sizeof(float), stream);
    hipMemsetAsync(counts, 0, 32, stream);

    moe_router<<<(T + 7) / 8, 256, 0, stream>>>(x, gate_w, topk, counts, T);
    moe_scan<<<1, 32, 0, stream>>>(counts, offsets, cursor);
    moe_scatter<<<(R + 255) / 256, 256, 0, stream>>>(topk, cursor, row_token, R);

    const int mtiles = (R + BM - 1) / BM;    // 32 (covers worst-case one-expert routing)
    dim3 gUp(INTER_DIM / BN, mtiles, NUM_EXPERTS);
    moe_up<<<gUp, 256, 0, stream>>>(x, w_up, h, counts, offsets, row_token);
    dim3 gDn(MODEL_DIM / BN, mtiles, NUM_EXPERTS);
    moe_down<<<gDn, 256, 0, stream>>>(h, w_down, out, counts, offsets, row_token);
}
